// AttentionModel_39848706572868
// MI455X (gfx1250) — matrix operand order repaired and checked by reading, not compiled, not run on hardware
//
#include <hip/hip_runtime.h>
#include <math.h>

typedef __attribute__((ext_vector_type(16))) _Float16 v16h;
typedef __attribute__((ext_vector_type(8)))  _Float16 v8h;
typedef __attribute__((ext_vector_type(8)))  float    v8f;
typedef unsigned int u32x4 __attribute__((ext_vector_type(4)));
typedef int          i32x8 __attribute__((ext_vector_type(8)));
typedef int          i32x4 __attribute__((ext_vector_type(4)));

#define B_BATCH 2
#define SLEN    2048
#define NH      16
#define DH      64
#define BQ      16        // query rows per wave
#define WAVES   4         // waves per block -> 64 query rows per block
#define BKV     64        // key/value chunk
#define LDK     72        // LDS row stride in halves (TDM pad: +4 dwords per 32 dwords)
#define TILEH   (BKV * LDK)   // padded tile size in halves
#define NCHUNK  (SLEN / BKV)

#define KEEP_BYTE 230u            // ~0.9 * 256 keep probability per byte lane
#define INV_KEEP  (1.0f / 0.9f)   // inverted-dropout scale (folded in epilogue)
#define LOG2E     1.4426950408889634f

__device__ __forceinline__ unsigned hash_u32(unsigned x) {
    x ^= x >> 16; x *= 0x7feb352dU;
    x ^= x >> 15; x *= 0x846ca68bU;
    x ^= x >> 16;
    return x;
}

// Build a 16x32 A-fragment from two contiguous 16-byte LDS runs (+0, +16 halves)
__device__ __forceinline__ v16h load_a_frag(const _Float16* base) {
    const v8h lo = *(const v8h*)(base);
    const v8h hi = *(const v8h*)(base + 16);
    v16h a;
    #pragma unroll
    for (int j = 0; j < 8; ++j) { a[j] = lo[j]; a[8 + j] = hi[j]; }
    return a;
}

// ---- TDM: DMA one 64x64 f16 tile (row = 64 elems, row stride = stride_el elems)
// into LDS at lds_addr, padding 4 dwords after every 32 dwords (-> LDK=72 rows).
__device__ __forceinline__ void tdm_load_tile(const _Float16* gsrc, unsigned lds_addr,
                                              unsigned stride_el, unsigned dim0_el) {
    const unsigned long long ga = (unsigned long long)(const void*)gsrc;
    u32x4 g0;
    g0[0] = 1u;                                     // count=1, user mode
    g0[1] = lds_addr;                               // LDS byte address
    g0[2] = (unsigned)ga;                           // global_addr[31:0]
    g0[3] = (unsigned)(ga >> 32) | 0x80000000u;     // global_addr[56:32] | type=2
    i32x8 g1;
    g1[0] = 0x07110000;     // data_size=2B | pad_enable | pad_interval=32dw | pad_amount=4dw
    g1[1] = (int)(dim0_el << 16);                   // tensor_dim0[15:0]
    g1[2] = (int)((64u << 16) | (dim0_el >> 16));   // tensor_dim1=64 | tensor_dim0[31:16]
    g1[3] = (int)(64u << 16);                       // tile_dim0=64
    g1[4] = 64;                                     // tile_dim1=64, tile_dim2=0
    g1[5] = (int)stride_el;                         // tensor_dim0_stride
    g1[6] = 0;
    g1[7] = 0;
    const i32x4 z4 = {0, 0, 0, 0};
    const i32x8 z8 = {0, 0, 0, 0, 0, 0, 0, 0};
    __builtin_amdgcn_tensor_load_to_lds(g0, g1, z4, z4, z8, 0);
}

// ---- pre-pass 1: Kh[b,h,n,d] = (f16) K[b,n,h,d]   (coalesced both sides) ----
__global__ __launch_bounds__(256)
void repack_k(const float* __restrict__ K, _Float16* __restrict__ Kh) {
    const int i  = blockIdx.x * 256 + threadIdx.x;   // B*NH*SLEN*(DH/4) threads
    const int d4 = i & (DH / 4 - 1);
    const int n  = (i >> 4) & (SLEN - 1);
    const int bh = i >> 15;
    const int b = bh >> 4, h = bh & (NH - 1);
    const float4 f = *(const float4*)(K + ((size_t)(b * SLEN + n) * NH + h) * DH + d4 * 4);
    union { _Float16 h4[4]; unsigned long long u; } p;
    p.h4[0] = (_Float16)f.x; p.h4[1] = (_Float16)f.y;
    p.h4[2] = (_Float16)f.z; p.h4[3] = (_Float16)f.w;
    *(unsigned long long*)(Kh + ((size_t)bh * SLEN + n) * DH + d4 * 4) = p.u;
}

// ---- pre-pass 2: Vt[b,h,d,n] = (f16) V[b,n,h,d]   (transpose; coalesced writes) ----
__global__ __launch_bounds__(256)
void repack_v(const float* __restrict__ V, _Float16* __restrict__ Vt) {
    const int i  = blockIdx.x * 256 + threadIdx.x;   // B*NH*DH*(SLEN/4) threads
    const int n4 = i & (SLEN / 4 - 1);
    const int d  = (i >> 9) & (DH - 1);
    const int bh = i >> 15;
    const int b = bh >> 4, h = bh & (NH - 1);
    const int n0 = n4 * 4;
    union { _Float16 h4[4]; unsigned long long u; } p;
    #pragma unroll
    for (int k = 0; k < 4; ++k)
        p.h4[k] = (_Float16)V[((size_t)(b * SLEN + n0 + k) * NH + h) * DH + d];
    *(unsigned long long*)(Vt + ((size_t)bh * DH + d) * SLEN + n0) = p.u;
}

__global__ __launch_bounds__(WAVES * 32)
void fa_fwd_f16wmma(const float* __restrict__ Q, const _Float16* __restrict__ Kh,
                    const _Float16* __restrict__ Vt, const float* __restrict__ inv_scale,
                    float* __restrict__ O) {
    __shared__ _Float16 sK [2][TILEH];   // double-buffered K chunk, [key][d], stride LDK
    __shared__ _Float16 sVt[2][TILEH];   // double-buffered V^T chunk, [d][key]

    const int tid  = threadIdx.x;
    const int wave = tid >> 5;
    const int lane = tid & 31;
    const int hl   = (lane >> 4) & 1;   // which 16-lane half
    const int ln   = lane & 15;

    const int b  = blockIdx.z;
    const int h  = blockIdx.y;
    const int bh = b * NH + h;
    const int q0 = blockIdx.x * (WAVES * BQ) + wave * BQ;

    const size_t bh_off = ((size_t)b * SLEN * NH + (size_t)h) * DH;  // fp32 [B,S,H,D]
    const _Float16* Ktile0 = Kh + (size_t)bh * SLEN * DH;            // + t0*DH per chunk
    const _Float16* Vtile0 = Vt + (size_t)bh * DH * SLEN;            // + t0 per chunk
    // fold softmax scale AND log2(e) into Q so each prob is a single raw v_exp_f32
    const float scl = LOG2E / inv_scale[0];

    // ---- Q^T B-fragments: lane ln owns query q0+ln; table-order d per frag ----
    v16h bQ[2];
    {
        const float* qrow = Q + bh_off + (size_t)(q0 + ln) * (NH * DH);
        #pragma unroll
        for (int c = 0; c < 2; ++c) {
            const int d0 = c * 32 + hl * 8;
            #pragma unroll
            for (int e = 0; e < 16; e += 4) {
                const float4 f = *(const float4*)(qrow + d0 + e + (e & 8));
                bQ[c][e + 0] = (_Float16)(f.x * scl);
                bQ[c][e + 1] = (_Float16)(f.y * scl);
                bQ[c][e + 2] = (_Float16)(f.z * scl);
                bQ[c][e + 3] = (_Float16)(f.w * scl);
            }
        }
    }

    v8f oacc[4];
    const v8f vzero = {0.f, 0.f, 0.f, 0.f, 0.f, 0.f, 0.f, 0.f};
    #pragma unroll
    for (int dt = 0; dt < 4; ++dt) oacc[dt] = vzero;
    float lpart = 0.f;   // this lane's partial softmax denominator (query = ln)

    const unsigned qkey = (unsigned)(bh * SLEN + q0 + ln);

    // ---- prologue: wave 0 DMAs chunk 0 into buffer 0 ----
    if (wave == 0) {
        tdm_load_tile(Ktile0, (unsigned)(size_t)&sK[0][0],  DH,   DH);
        tdm_load_tile(Vtile0, (unsigned)(size_t)&sVt[0][0], SLEN, SLEN);
    }

    #pragma unroll 1
    for (int kc = 0; kc < NCHUNK; ++kc) {
        const int t0  = kc * BKV;
        const int buf = kc & 1;
        if (wave == 0) __builtin_amdgcn_s_wait_tensorcnt(0);
        __syncthreads();   // current buf ready for all; other buf fully consumed
        if (wave == 0 && kc + 1 < NCHUNK) {   // prefetch next chunk via TDM
            tdm_load_tile(Ktile0 + (size_t)(t0 + BKV) * DH,
                          (unsigned)(size_t)&sK[buf ^ 1][0],  DH,   DH);
            tdm_load_tile(Vtile0 + (size_t)(t0 + BKV),
                          (unsigned)(size_t)&sVt[buf ^ 1][0], SLEN, SLEN);
        }

        // ---- S^T = K Q^T: C-layout => lane = query (ln), register = key ----
        v8f sacc[4];
        #pragma unroll
        for (int nt = 0; nt < 4; ++nt) {
            sacc[nt] = vzero;
            #pragma unroll
            for (int c = 0; c < 2; ++c) {
                const v16h aK =
                    load_a_frag(&sK[buf][(nt * 16 + ln) * LDK + c * 32 + hl * 8]);
                sacc[nt] = __builtin_amdgcn_wmma_f32_16x16x32_f16(
                    false, aK, false, bQ[c], (short)0, sacc[nt], false, false);
            }
        }

        // ---- raw-exp2 + dropout; pack P straight into A-fragments (no LDS/shfl) ----
        v16h aP[2];
        #pragma unroll
        for (int nt = 0; nt < 4; ++nt) {
            const unsigned nbase = (unsigned)(t0 + nt * 16 + hl * 8);
            #pragma unroll
            for (int rg = 0; rg < 2; ++rg) {
                const unsigned hsh = hash_u32(qkey * 0x9E3779B1u +
                                              ((nbase >> 2) + (unsigned)rg) * 0x85EBCA77u);
                #pragma unroll
                for (int k = 0; k < 4; ++k) {
                    const int r = rg * 4 + k;
                    const float p = __builtin_amdgcn_exp2f(sacc[nt][r]);
                    lpart += p;                            // denominator: un-dropped sum
                    const bool keep = ((hsh >> (k * 8)) & 255u) < KEEP_BYTE;
                    aP[nt >> 1][(nt & 1) * 8 + r] = (_Float16)(keep ? p : 0.f);
                }
            }
        }

        // ---- O += P * V: B-fragments of V^T are contiguous LDS runs ----
        #pragma unroll
        for (int c = 0; c < 2; ++c) {
            #pragma unroll
            for (int dt = 0; dt < 4; ++dt) {
                const v16h bV =
                    load_a_frag(&sVt[buf][(dt * 16 + ln) * LDK + c * 32 + hl * 8]);
                oacc[dt] = __builtin_amdgcn_wmma_f32_16x16x32_f16(
                    false, aP[c], false, bV, (short)0, oacc[dt], false, false);
            }
        }
        __syncthreads();   // all reads of buf done before it becomes the DMA target
    }

    // ---- one-shot denominator reduction + normalize (1/keep folded) + store ----
    const float lsum = lpart + __shfl_xor(lpart, 16, 32);  // lane ln: l(query ln)
    const size_t obase = (((size_t)b * NH + h) * SLEN) * DH;
    #pragma unroll
    for (int r = 0; r < 8; ++r) {
        const float lq = __shfl(lsum, hl * 8 + r, 32);     // l for this C-row's query
        const float linv = INV_KEEP * __builtin_amdgcn_rcpf(lq);
        const size_t rowo = obase + (size_t)(q0 + hl * 8 + r) * DH;
        #pragma unroll
        for (int dt = 0; dt < 4; ++dt)
            O[rowo + dt * 16 + ln] = oacc[dt][r] * linv;
    }
}

extern "C" void kernel_launch(void* const* d_in, const int* in_sizes, int n_in,
                              void* d_out, int out_size, void* d_ws, size_t ws_size,
                              hipStream_t stream) {
    (void)in_sizes; (void)n_in; (void)out_size; (void)ws_size;
    const float* Q  = (const float*)d_in[0];
    const float* K  = (const float*)d_in[1];
    const float* V  = (const float*)d_in[2];
    const float* is = (const float*)d_in[3];
    float* O = (float*)d_out;

    _Float16* Kh = (_Float16*)d_ws;                                   // 8 MB
    _Float16* Vt = Kh + (size_t)B_BATCH * NH * SLEN * DH;             // 8 MB

    const int nconv = B_BATCH * NH * SLEN * (DH / 4);                 // 1M threads
    repack_k<<<nconv / 256, 256, 0, stream>>>(K, Kh);
    repack_v<<<nconv / 256, 256, 0, stream>>>(V, Vt);

    dim3 grid(SLEN / (WAVES * BQ), NH, B_BATCH);
    dim3 block(WAVES * 32);
    fa_fwd_f16wmma<<<grid, block, 0, stream>>>(Q, Kh, Vt, is, O);
}
